// GeneralAttention_20083267076703
// MI455X (gfx1250) — matrix operand order repaired and checked by reading, not compiled, not run on hardware
//
#include <hip/hip_runtime.h>
#include <math.h>

// ---------------- problem constants ----------------
#define BB 8
#define QQ 128
#define KK 512
#define DD 512     // D_Q == D_K == D_V
#define HH 256
#define VV 512

typedef __bf16 bf16_t;
typedef __attribute__((ext_vector_type(8)))  __bf16 v8bf;
typedef __attribute__((ext_vector_type(16))) __bf16 v16bf;
typedef __attribute__((ext_vector_type(8)))  float  v8f;

// ---------------- ws layout (bytes) ----------------
#define OFF_QBF   0u                        // queries bf16       [B*Q, D]   1 MB
#define OFF_KBF   1048576u                  // keys bf16          [B*K, D]   4 MB
#define OFF_WQBF  5242880u                  // Wq bf16            [H, D]     256 KB
#define OFF_WKBF  5505024u                  // Wk bf16            [H, D]     256 KB
#define OFF_VTHI  5767168u                  // valuesT hi bf16    [B, V, K]  4 MB
#define OFF_VTLO  9961472u                  // valuesT lo bf16    [B, V, K]  4 MB
#define OFF_QIN   14155776u                 // q_in f32           [B*Q, H]   1 MB
#define OFF_KIN   15204352u                 // k_in f32           [B*K, H]   4 MB
#define OFF_AHI   19398656u                 // align hi bf16      [B*Q, K]   1 MB
#define OFF_ALO   20447232u                 // align lo bf16      [B*Q, K]   1 MB

// ---------------- WMMA fragment loaders ----------------
// A operand 16x32 bf16 (row-major source, ld = K elements):
//   lane l: row = l&15, khalf = l>>4; one contiguous 16-elem (32B) chunk at
//   k + khalf*16.
__device__ __forceinline__ v16bf load_a_frag(const bf16_t* row_ptr, int kbase, int khalf) {
  return *(const v16bf*)(row_ptr + kbase + khalf * 16);
}
// B operand 32x16 bf16 from a row-major B^T source (lane = output column n):
//   lane l: col = l&15, khalf = l>>4; one contiguous 16-elem (32B) chunk at
//   k + khalf*16.
__device__ __forceinline__ v16bf load_b_frag(const bf16_t* row_ptr, int kbase, int khalf) {
  return *(const v16bf*)(row_ptr + kbase + khalf * 16);
}
__device__ __forceinline__ v8f wmma_bf16(v16bf a, v16bf b, v8f c) {
  return __builtin_amdgcn_wmma_f32_16x16x32_bf16(false, a, false, b, (short)0, c, false, false);
}

// ---------------- prep kernels ----------------
__global__ void __launch_bounds__(256) cvt_bf16_kernel(const float* __restrict__ in,
                                                       bf16_t* __restrict__ out, int n) {
  int i = blockIdx.x * blockDim.x + threadIdx.x;
  if (i < n) out[i] = (bf16_t)in[i];
}

// values [B,K,V] f32 -> valuesT hi/lo bf16 [B,V,K]
__global__ void __launch_bounds__(256) vtrans_split_kernel(const float* __restrict__ vals,
                                                           bf16_t* __restrict__ hi,
                                                           bf16_t* __restrict__ lo) {
  int idx = blockIdx.x * blockDim.x + threadIdx.x;      // over B*V*K
  if (idx >= BB * VV * KK) return;
  int k = idx & (KK - 1);
  int v = (idx >> 9) & (VV - 1);
  int b = idx >> 18;
  float x = vals[((size_t)b * KK + k) * VV + v];
  bf16_t h = (bf16_t)x;
  hi[idx] = h;
  lo[idx] = (bf16_t)(x - (float)h);
}

// ---------------- phase 1: X @ W^T + bias (bf16 WMMA, f32 acc) ----------------
__global__ void __launch_bounds__(256) gemm_nt_bias_kernel(
    const bf16_t* __restrict__ A,    // [M, Kdim] bf16 row-major
    const bf16_t* __restrict__ Bt,   // [N, Kdim] bf16 row-major (B transposed)
    const float*  __restrict__ bias, // [N]
    float*        __restrict__ C,    // [M, N] f32
    int Mtiles, int Ntiles, int Kdim, int N) {
  int wave = threadIdx.x >> 5;
  int lane = threadIdx.x & 31;
  int tile = blockIdx.x * (blockDim.x >> 5) + wave;
  if (tile >= Mtiles * Ntiles) return;                  // wave-uniform
  int tm = tile / Ntiles, tn = tile % Ntiles;
  int lrow = lane & 15, khalf = lane >> 4;
  const bf16_t* arow = A  + (size_t)(tm * 16 + lrow) * Kdim;
  const bf16_t* brow = Bt + (size_t)(tn * 16 + lrow) * Kdim;
  v8f acc = {};
  for (int k = 0; k < Kdim; k += 32) {
    v16bf af = load_a_frag(arow, k, khalf);
    v16bf bf = load_b_frag(brow, k, khalf);
    acc = wmma_bf16(af, bf, acc);
  }
  int col   = tn * 16 + lrow;
  int rbase = tm * 16 + khalf * 8;
  float bv = bias[col];
#pragma unroll
  for (int j = 0; j < 8; ++j)
    C[(size_t)(rbase + j) * N + col] = acc[j] + bv;
}

// ---------------- phase 2: fused tanh-score + mask + softmax ----------------
// one block (8 waves) per (b,q) row; wave handles one k at a time,
// lanes parallelize h with a wave32 shuffle reduction.
__global__ void __launch_bounds__(256) scores_softmax_kernel(
    const float* __restrict__ q_in,   // [B*Q, H]
    const float* __restrict__ k_in,   // [B*K, H]
    const float* __restrict__ v_a,    // [H]
    const int*   __restrict__ bank_length,  // [B]
    const int*   __restrict__ vbs_ptr,      // [1]
    float*  __restrict__ align_out,   // [B*Q, K] f32 (second output)
    bf16_t* __restrict__ align_hi,    // [B*Q, K]
    bf16_t* __restrict__ align_lo) {
  __shared__ float qrow[HH];
  __shared__ float va[HH];
  __shared__ float sc[KK];
  __shared__ float rbuf[256];

  const int row = blockIdx.x;        // b*Q + q
  const int b   = row >> 7;
  const int t   = threadIdx.x;
  qrow[t] = q_in[(size_t)row * HH + t];
  va[t]   = v_a[t];
  __syncthreads();

  const int blen = bank_length[b];
  const int vbs  = vbs_ptr[0];
  const int wave = t >> 5, lane = t & 31;
  const float* kin_b = k_in + (size_t)b * KK * HH;

  for (int k = wave; k < KK; k += 8) {
    float s;
    if (k >= blen || (vbs > 0 && k == 0)) {
      s = -__builtin_inff();
    } else {
      const float* kr = kin_b + (size_t)k * HH;
      float acc = 0.0f;
#pragma unroll
      for (int h = lane; h < HH; h += 32)
        acc += va[h] * tanhf(qrow[h] + kr[h]);
#pragma unroll
      for (int off = 16; off > 0; off >>= 1)
        acc += __shfl_xor(acc, off, 32);
      s = acc;
    }
    if (lane == 0) sc[k] = s;
  }
  __syncthreads();

  // block softmax over sc[0..511]
  rbuf[t] = fmaxf(sc[t], sc[t + 256]);
  __syncthreads();
  for (int s2 = 128; s2 > 0; s2 >>= 1) {
    if (t < s2) rbuf[t] = fmaxf(rbuf[t], rbuf[t + s2]);
    __syncthreads();
  }
  const float mx = rbuf[0];
  __syncthreads();

  const float e0 = expf(sc[t] - mx);
  const float e1 = expf(sc[t + 256] - mx);
  rbuf[t] = e0 + e1;
  __syncthreads();
  for (int s2 = 128; s2 > 0; s2 >>= 1) {
    if (t < s2) rbuf[t] += rbuf[t + s2];
    __syncthreads();
  }
  const float inv = 1.0f / rbuf[0];

  const float a0 = e0 * inv, a1 = e1 * inv;
  const size_t base = (size_t)row * KK;
  align_out[base + t]       = a0;
  align_out[base + t + 256] = a1;
  bf16_t h0 = (bf16_t)a0, h1 = (bf16_t)a1;
  align_hi[base + t]        = h0;
  align_hi[base + t + 256]  = h1;
  align_lo[base + t]        = (bf16_t)(a0 - (float)h0);
  align_lo[base + t + 256]  = (bf16_t)(a1 - (float)h1);
}

// ---------------- phase 3: out = align @ values (split-bf16 WMMA) ----------------
__global__ void __launch_bounds__(256) av_gemm_kernel(
    const bf16_t* __restrict__ Ahi, const bf16_t* __restrict__ Alo, // [B*Q, K]
    const bf16_t* __restrict__ Bhi, const bf16_t* __restrict__ Blo, // [B, V, K]
    float* __restrict__ out) {                                      // [B, Q, V]
  int wave = threadIdx.x >> 5;
  int lane = threadIdx.x & 31;
  int tile = blockIdx.x * 8 + wave;        // 2048 wave-tiles total
  int b    = tile >> 8;                    // 8 Mtiles * 32 Ntiles per batch
  int rem  = tile & 255;
  int tm = rem >> 5, tn = rem & 31;
  int lrow = lane & 15, khalf = lane >> 4;

  const bf16_t* ah = Ahi + ((size_t)b * QQ + tm * 16 + lrow) * KK;
  const bf16_t* al = Alo + ((size_t)b * QQ + tm * 16 + lrow) * KK;
  const bf16_t* bh = Bhi + ((size_t)b * VV + tn * 16 + lrow) * KK;
  const bf16_t* bl = Blo + ((size_t)b * VV + tn * 16 + lrow) * KK;

  v8f acc = {};
  for (int k = 0; k < KK; k += 32) {
    v16bf a0 = load_a_frag(ah, k, khalf);
    v16bf a1 = load_a_frag(al, k, khalf);
    v16bf b0 = load_b_frag(bh, k, khalf);
    v16bf b1 = load_b_frag(bl, k, khalf);
    acc = wmma_bf16(a0, b0, acc);   // hi*hi
    acc = wmma_bf16(a0, b1, acc);   // hi*lo
    acc = wmma_bf16(a1, b0, acc);   // lo*hi
  }
  float* outb = out + (size_t)b * QQ * VV;
  int col = tn * 16 + lrow;
  int rbase = tm * 16 + khalf * 8;
#pragma unroll
  for (int j = 0; j < 8; ++j)
    outb[(size_t)(rbase + j) * VV + col] = acc[j];
}

// ---------------- launcher ----------------
extern "C" void kernel_launch(void* const* d_in, const int* in_sizes, int n_in,
                              void* d_out, int out_size, void* d_ws, size_t ws_size,
                              hipStream_t stream) {
  const float* queries = (const float*)d_in[0];   // [B,Q,D]
  const float* keys    = (const float*)d_in[1];   // [B,K,D]
  const float* values  = (const float*)d_in[2];   // [B,K,V]
  const int*   blen    = (const int*)  d_in[3];   // [B]
  const int*   vbs     = (const int*)  d_in[4];   // [1]
  const float* Wq      = (const float*)d_in[5];   // [H,D]
  const float* bq      = (const float*)d_in[6];   // [H]
  const float* Wk      = (const float*)d_in[7];   // [H,D]
  const float* bk      = (const float*)d_in[8];   // [H]
  const float* v_a     = (const float*)d_in[9];   // [H]

  char* ws = (char*)d_ws;
  bf16_t* qbf  = (bf16_t*)(ws + OFF_QBF);
  bf16_t* kbf  = (bf16_t*)(ws + OFF_KBF);
  bf16_t* wqbf = (bf16_t*)(ws + OFF_WQBF);
  bf16_t* wkbf = (bf16_t*)(ws + OFF_WKBF);
  bf16_t* vthi = (bf16_t*)(ws + OFF_VTHI);
  bf16_t* vtlo = (bf16_t*)(ws + OFF_VTLO);
  float*  q_in = (float*) (ws + OFF_QIN);
  float*  k_in = (float*) (ws + OFF_KIN);
  bf16_t* ahi  = (bf16_t*)(ws + OFF_AHI);
  bf16_t* alo  = (bf16_t*)(ws + OFF_ALO);

  float* out_av    = (float*)d_out;                      // [B,Q,V]
  float* out_align = (float*)d_out + (size_t)BB*QQ*VV;   // [B,Q,K]

  // prep: f32 -> bf16 planes (+ transposed split values)
  cvt_bf16_kernel<<<(BB*QQ*DD)/256, 256, 0, stream>>>(queries, qbf, BB*QQ*DD);
  cvt_bf16_kernel<<<(BB*KK*DD)/256, 256, 0, stream>>>(keys,    kbf, BB*KK*DD);
  cvt_bf16_kernel<<<(HH*DD)/256,    256, 0, stream>>>(Wq,     wqbf, HH*DD);
  cvt_bf16_kernel<<<(HH*DD)/256,    256, 0, stream>>>(Wk,     wkbf, HH*DD);
  vtrans_split_kernel<<<(BB*VV*KK)/256, 256, 0, stream>>>(values, vthi, vtlo);

  // phase 1: projections (WMMA)
  gemm_nt_bias_kernel<<<(64*16)/8,  256, 0, stream>>>(qbf, wqbf, bq, q_in,  64, 16, DD, HH);
  gemm_nt_bias_kernel<<<(256*16)/8, 256, 0, stream>>>(kbf, wkbf, bk, k_in, 256, 16, DD, HH);

  // phase 2: fused tanh scores + mask + softmax
  scores_softmax_kernel<<<BB*QQ, 256, 0, stream>>>(q_in, k_in, v_a, blen, vbs,
                                                   out_align, ahi, alo);

  // phase 3: align @ values (split-bf16 WMMA)
  av_gemm_kernel<<<(BB*8*32)/8, 256, 0, stream>>>(ahi, alo, vthi, vtlo, out_av);
}
